// SelfAttention_25984552141279
// MI455X (gfx1250) — matrix operand order repaired and checked by reading, not compiled, not run on hardware
//
#include <hip/hip_runtime.h>

#define BATCH 16
#define SEQ   2304
#define DM    256
#define SCALE 0.0625f   // 1/sqrt(256)

#define USE_ASYNC_LDS 1

typedef _Float16 v16h __attribute__((ext_vector_type(16)));
typedef _Float16 h8   __attribute__((ext_vector_type(8)));
typedef float    v8f  __attribute__((ext_vector_type(8)));
typedef float    f4   __attribute__((ext_vector_type(4)));

static __device__ inline v16h mk16(h8 lo, h8 hi) {
  v16h r;
#pragma unroll
  for (int i = 0; i < 8; ++i) { r[i] = lo[i]; r[i + 8] = hi[i]; }
  return r;
}
static __device__ inline v8f vzero8() {
  v8f r;
#pragma unroll
  for (int i = 0; i < 8; ++i) r[i] = 0.0f;
  return r;
}
// B fragment from a row-base pointer: element e -> K-dim = kk*32 + hf*16 + e
static __device__ inline v16h ldB(const _Float16* base, int kk, int hf) {
  const _Float16* p = base + kk * 32 + hf * 16;
  return mk16(*(const h8*)p, *(const h8*)(p + 8));
}
// reduce over the 16 lanes of a half-wave (xor masks < 16 never cross halves)
static __device__ inline float rmax16(float v) {
#pragma unroll
  for (int m = 1; m < 16; m <<= 1) v = fmaxf(v, __shfl_xor(v, m, 32));
  return v;
}
static __device__ inline float rsum16(float v) {
#pragma unroll
  for (int m = 1; m < 16; m <<= 1) v += __shfl_xor(v, m, 32);
  return v;
}

// CDNA5 async global->LDS copy (ASYNCcnt-tracked, no VGPR round trip).
static __device__ inline void gload_async_b128(void* lds, const void* g) {
#if USE_ASYNC_LDS
  uint32_t l = (uint32_t)(uintptr_t)lds;
  uint64_t a = (uint64_t)(uintptr_t)g;
  asm volatile("global_load_async_to_lds_b128 %0, %1, off"
               :: "v"(l), "v"(a) : "memory");
#else
  *(uint4*)lds = *(const uint4*)g;
#endif
}
static __device__ inline void wait_async_lds() {
#if USE_ASYNC_LDS
  asm volatile("s_wait_asynccnt 0x0" ::: "memory");
#endif
}

// ---------------------------------------------------------------------------
// Kernel 0: one-time weight repack: Wt[mat][n][d] = f16(w_mat[d][n]).
// ---------------------------------------------------------------------------
__global__ __launch_bounds__(256) void wprep_kernel(
    const float* __restrict__ wq, const float* __restrict__ wk,
    const float* __restrict__ wv, _Float16* __restrict__ Wt)
{
  int idx = blockIdx.x * 256 + threadIdx.x;     // 3*256*256 total
  int mat = idx >> 16;
  int r   = idx & 65535;
  int n   = r >> 8;
  int d   = r & 255;
  const float* w = (mat == 0) ? wq : (mat == 1) ? wk : wv;
  Wt[idx] = (_Float16)w[d * DM + n];
}

// ---------------------------------------------------------------------------
// Kernel 1: fused (x + pos_emb) -> Q,K,V projections in f16.
// Workgroup = 128 rows. Phase 1: stage f16(x+pos) in LDS, lift per-wave
// A-fragments into VGPRs. Phase 2: reuse the 64KB LDS for each 256x256 f16
// weight matrix (async-staged); waves sweep 2 column tiles at a time with
// dual accumulator chains and B-fragment prefetch.
// ---------------------------------------------------------------------------
__global__ __launch_bounds__(256) void qkv_kernel(
    const float* __restrict__ x, const float* __restrict__ pos,
    const _Float16* __restrict__ Wt,
    const float* __restrict__ bq, const float* __restrict__ bk,
    const float* __restrict__ bv,
    _Float16* __restrict__ Qm, _Float16* __restrict__ Km,
    _Float16* __restrict__ Vt)
{
  __shared__ __align__(16) _Float16 sbuf[128 * DM];   // 64 KB (xh, then W)
  uint4* sbuf4 = (uint4*)sbuf;

  int tid     = threadIdx.x;
  int rowBase = blockIdx.x * 128;

  // ---- phase 1a: stage f16(x + pos) : 2 threads per row, 128 d each ----
  {
    int row  = tid >> 1;
    int d0   = (tid & 1) * 128;
    int rowG = rowBase + row;
    int s    = rowG % SEQ;
    const float* xr = x   + (size_t)rowG * DM + d0;
    const float* pr = pos + (size_t)s    * DM + d0;
    _Float16* dst = sbuf + row * DM + d0;
#pragma unroll
    for (int j = 0; j < 128; j += 8) {
      f4 a0 = *(const f4*)(xr + j)     + *(const f4*)(pr + j);
      f4 a1 = *(const f4*)(xr + j + 4) + *(const f4*)(pr + j + 4);
      h8 o;
#pragma unroll
      for (int i = 0; i < 4; ++i) {
        o[i]     = (_Float16)a0[i];
        o[i + 4] = (_Float16)a1[i];
      }
      *(h8*)(dst + j) = o;
    }
  }
  __syncthreads();

  int wv_  = tid >> 5;
  int lane = tid & 31;
  int hf   = lane >> 4;
  int l16  = lane & 15;
  int kb   = hf * 8;

  // ---- phase 1b: resident A fragments for this wave's 16 rows ----
  v16h af[8];
  {
    const _Float16* ar = sbuf + (wv_ * 16 + l16) * DM;
#pragma unroll
    for (int kk = 0; kk < 8; ++kk)
      af[kk] = mk16(*(const h8*)(ar + kk * 32 + 2 * kb),
                    *(const h8*)(ar + kk * 32 + 2 * kb + 8));
  }
  __syncthreads();          // everyone done reading x-halves; sbuf reusable

  int rowTile = rowBase + wv_ * 16;     // global first row of this tile
  int b       = rowTile / SEQ;          // 128-row blocks never straddle a batch
  int sBase   = rowTile % SEQ + hf * 8;

  for (int mat = 0; mat < 3; ++mat) {
    // ---- phase 2a: async-stage n-rows 0..127 (64 KB) of this matrix's f16 weights ----
    const uint4* wsrc = (const uint4*)(Wt + ((size_t)mat << 16));
#pragma unroll
    for (int j = 0; j < 16; ++j)
      gload_async_b128(&sbuf4[tid + j * 256], wsrc + tid + j * 256);
    wait_async_lds();
    __syncthreads();

    const float* bias = (mat == 0) ? bq : (mat == 1) ? bk : bv;

    auto store_tile = [&](v8f c, int n) {
      float bn = bias[n];
      if (mat < 2) {                     // Q/K row-major
        _Float16* dst = ((mat == 0) ? Qm : Km) +
                        (size_t)(rowTile + hf * 8) * DM + n;
#pragma unroll
        for (int r = 0; r < 8; ++r)
          dst[(size_t)r * DM] = (_Float16)(c[r] + bn);
      } else {                           // V transposed: Vt[b][n][s]
        _Float16* dst = Vt + (size_t)b * DM * SEQ + (size_t)n * SEQ + sBase;
#pragma unroll
        for (int r = 0; r < 8; ++r) dst[r] = (_Float16)(c[r] + bn);
      }
    };

    // ---- phase 2b: two column tiles per pass, prefetched B fragments ----
    for (int ct = 0; ct < 16; ct += 2) {
      int n0 = ct * 16 + l16;
      int n1 = n0 + 16;
      const _Float16* w0 = sbuf + (size_t)n0 * DM;
      const _Float16* w1 = sbuf + (size_t)n1 * DM;
      v8f c0 = vzero8(), c1 = vzero8();
      v16h B0 = ldB(w0, 0, hf);
      v16h B1 = ldB(w1, 0, hf);
#pragma unroll
      for (int kk = 0; kk < 8; ++kk) {
        v16h nB0 = B0, nB1 = B1;
        if (kk < 7) {                    // prefetch next k-step before wmma
          nB0 = ldB(w0, kk + 1, hf);
          nB1 = ldB(w1, kk + 1, hf);
        }
        c0 = __builtin_amdgcn_wmma_f32_16x16x32_f16(false, af[kk], false, B0,
                                                    (short)0, c0, false, false);
        c1 = __builtin_amdgcn_wmma_f32_16x16x32_f16(false, af[kk], false, B1,
                                                    (short)0, c1, false, false);
        B0 = nB0; B1 = nB1;
      }
      store_tile(c0, n0);
      store_tile(c1, n1);
    }
    __syncthreads();        // done reading weights before next mat overwrites
  }
}

// ---------------------------------------------------------------------------
// Kernel 2: flash attention. 8 waves/block, each owns a 16-query tile;
// block stages 32-key K / V^T blocks in LDS via async global->LDS copies.
// ---------------------------------------------------------------------------
__global__ __launch_bounds__(256, 1) void attn_kernel(
    const _Float16* __restrict__ Qm, const _Float16* __restrict__ Km,
    const _Float16* __restrict__ Vt, float* __restrict__ out)
{
  __shared__ uint4 Ks4[1024];                    // 32 keys x 256 d   (16 KB)
  __shared__ uint4 Vs4[1024];                    // 256 d   x 32 keys (16 KB)
  __shared__ __align__(16) _Float16 Ps[8 * 16 * 32];  // per-wave P staging (8 KB)

  _Float16* Ks = (_Float16*)Ks4;
  _Float16* Vs = (_Float16*)Vs4;

  int tid  = threadIdx.x;
  int wv_  = tid >> 5;
  int lane = tid & 31;
  int hf   = lane >> 4;
  int l16  = lane & 15;
  int kb   = hf * 8;

  int b      = blockIdx.x / (SEQ / 128);         // 18 q-groups per batch
  int qGroup = blockIdx.x % (SEQ / 128);
  int q0     = (qGroup * 8 + wv_) * 16;

  _Float16* Pw = Ps + wv_ * 512;

  // Q tile as 8 resident A fragments (k = 32 each)
  v16h qf[8];
  const _Float16* qrow = Qm + ((size_t)b * SEQ + q0 + l16) * DM;
#pragma unroll
  for (int kk = 0; kk < 8; ++kk) {
    h8 lo = *(const h8*)(qrow + kk * 32 + 2 * kb);
    h8 hi = *(const h8*)(qrow + kk * 32 + 2 * kb + 8);
    qf[kk] = mk16(lo, hi);
  }

  v8f acc[16];
#pragma unroll
  for (int nt = 0; nt < 16; ++nt) acc[nt] = vzero8();
  float mrow[8], lrow[8];
#pragma unroll
  for (int r = 0; r < 8; ++r) { mrow[r] = -3.0e38f; lrow[r] = 0.0f; }

  const uint4*    ksrc_base = (const uint4*)(Km + (size_t)b * SEQ * DM);
  const _Float16* vbase     = Vt + (size_t)b * DM * SEQ;

  for (int s0 = 0; s0 < SEQ; s0 += 32) {
    __syncthreads();                             // protect LDS reuse
    {                                            // async K/V staging
      const uint4* ksrc = ksrc_base + (size_t)s0 * (DM / 8);
#pragma unroll
      for (int j = 0; j < 4; ++j)
        gload_async_b128(&Ks4[tid + j * 256], ksrc + tid + j * 256);
      const uint4* vsrc = (const uint4*)(vbase + (size_t)tid * SEQ + s0);
#pragma unroll
      for (int j = 0; j < 4; ++j)
        gload_async_b128(&Vs4[tid * 4 + j], vsrc + j);
      wait_async_lds();
    }
    __syncthreads();

    // logits for two 16-key tiles: S = Q * K^T (dual chains + prefetch)
    v8f sA = vzero8(), sB = vzero8();
    {
      const _Float16* krow0 = Ks + (size_t)l16 * DM;
      const _Float16* krow1 = Ks + (size_t)(16 + l16) * DM;
      v16h B0 = ldB(krow0, 0, hf);
      v16h B1 = ldB(krow1, 0, hf);
#pragma unroll
      for (int kk = 0; kk < 8; ++kk) {
        v16h nB0 = B0, nB1 = B1;
        if (kk < 7) {
          nB0 = ldB(krow0, kk + 1, hf);
          nB1 = ldB(krow1, kk + 1, hf);
        }
        sA = __builtin_amdgcn_wmma_f32_16x16x32_f16(false, qf[kk], false, B0,
                                                    (short)0, sA, false, false);
        sB = __builtin_amdgcn_wmma_f32_16x16x32_f16(false, qf[kk], false, B1,
                                                    (short)0, sB, false, false);
        B0 = nB0; B1 = nB1;
      }
    }

    // online softmax (row stats live in the 16-lane half that owns each row)
#pragma unroll
    for (int r = 0; r < 8; ++r) {
      float v0   = sA[r] * SCALE;
      float v1   = sB[r] * SCALE;
      float mx   = rmax16(fmaxf(v0, v1));
      float mnew = fmaxf(mrow[r], mx);
      float corr = __expf(mrow[r] - mnew);
      float p0   = __expf(v0 - mnew);
      float p1   = __expf(v1 - mnew);
      float ls   = rsum16(p0 + p1);
      lrow[r] = lrow[r] * corr + ls;
      mrow[r] = mnew;
#pragma unroll
      for (int nt = 0; nt < 16; ++nt) acc[nt][r] *= corr;
      int m = hf * 8 + r;                        // C layout -> P[m][n]
      Pw[m * 32 + l16]      = (_Float16)p0;
      Pw[m * 32 + 16 + l16] = (_Float16)p1;
    }
    // wave-internal LDS RAW: wait for our own ds_stores, no block barrier needed
    asm volatile("s_wait_dscnt 0x0" ::: "memory");

    // A fragment of P (16 q x 32 keys), then acc += P * V (16 indep chains)
    {
      const _Float16* pr = Pw + (size_t)l16 * 32;
      v16h pA = mk16(*(const h8*)(pr + 2 * kb), *(const h8*)(pr + 2 * kb + 8));
      const _Float16* vb0 = Vs + (size_t)l16 * 32 + hf * 16;
      v16h Bf = mk16(*(const h8*)vb0, *(const h8*)(vb0 + 8));
#pragma unroll
      for (int nt = 0; nt < 16; ++nt) {
        v16h nBf = Bf;
        if (nt < 15) {                   // prefetch next d-tile's V fragment
          const _Float16* vp = Vs + (size_t)((nt + 1) * 16 + l16) * 32 + hf * 16;
          nBf = mk16(*(const h8*)vp, *(const h8*)(vp + 8));
        }
        acc[nt] = __builtin_amdgcn_wmma_f32_16x16x32_f16(false, pA, false, Bf,
                                                         (short)0, acc[nt],
                                                         false, false);
        Bf = nBf;
      }
    }
  }

  // normalize and store f32 output
#pragma unroll
  for (int r = 0; r < 8; ++r) {
    float inv = 1.0f / lrow[r];
    size_t rowOff = ((size_t)b * SEQ + q0 + hf * 8 + r) * DM;
#pragma unroll
    for (int nt = 0; nt < 16; ++nt)
      out[rowOff + nt * 16 + l16] = acc[nt][r] * inv;
  }
}

// ---------------------------------------------------------------------------
extern "C" void kernel_launch(void* const* d_in, const int* in_sizes, int n_in,
                              void* d_out, int out_size, void* d_ws, size_t ws_size,
                              hipStream_t stream) {
  const float* x   = (const float*)d_in[0];
  const float* wq  = (const float*)d_in[1];
  const float* bq  = (const float*)d_in[2];
  const float* wk  = (const float*)d_in[3];
  const float* bk  = (const float*)d_in[4];
  const float* wv  = (const float*)d_in[5];
  const float* bv  = (const float*)d_in[6];
  const float* pos = (const float*)d_in[7];

  size_t elems = (size_t)BATCH * SEQ * DM;
  _Float16* Qm = (_Float16*)d_ws;
  _Float16* Km = Qm + elems;
  _Float16* Vt = Km + elems;
  _Float16* Wt = Vt + elems;                      // 3*256*256 f16 (384 KB)

  wprep_kernel<<<3 * DM * DM / 256, 256, 0, stream>>>(wq, wk, wv, Wt);
  qkv_kernel<<<(BATCH * SEQ) / 128, 256, 0, stream>>>(x, pos, Wt, bq, bk, bv,
                                                      Qm, Km, Vt);
  attn_kernel<<<BATCH * (SEQ / 128), 256, 0, stream>>>(Qm, Km, Vt, (float*)d_out);
}
